// SelfAttention_44547400794204
// MI455X (gfx1250) — matrix operand order repaired and checked by reading, not compiled, not run on hardware
//
#include <hip/hip_runtime.h>

// ---------------------------------------------------------------------------
// MI455X (gfx1250) bf16-WMMA implementation of attention + FFN.
// Compute-bound (~515 GFLOP vs ~9us of HBM traffic) -> everything through
// v_wmma_f32_16x16x32_bf16, f32 accumulate, f32 softmax/bias/residual.
// Q-tile staging in the attention kernel uses the Tensor Data Mover
// (tensor_load_to_lds + s_wait_tensorcnt).
// ---------------------------------------------------------------------------

#define USE_TDM 1

typedef __bf16 bf16_t;
typedef __attribute__((ext_vector_type(8)))  __bf16 v8bf;
typedef __attribute__((ext_vector_type(16))) __bf16 v16bf;
typedef __attribute__((ext_vector_type(8)))  float  v8f;
typedef __attribute__((ext_vector_type(4)))  unsigned int u32x4;
typedef __attribute__((ext_vector_type(8)))  unsigned int u32x8;

#define BB 8
#define TT 2048
#define EE 1024
#define HH 4096
#define MM (BB * TT)   // 16384 rows

__device__ inline v8bf ld8(const bf16_t* p) {
    return *reinterpret_cast<const v8bf*>(p);
}
__device__ inline v16bf make_frag(v8bf lo, v8bf hi) {
    v16bf r;
#pragma unroll
    for (int i = 0; i < 8; ++i) { r[i] = lo[i]; r[i + 8] = hi[i]; }
    return r;
}
__device__ inline v8f vzero8() {
    v8f z;
#pragma unroll
    for (int i = 0; i < 8; ++i) z[i] = 0.0f;
    return z;
}
__device__ inline v8f wmma_bf16(v16bf a, v16bf b, v8f c) {
    return __builtin_amdgcn_wmma_f32_16x16x32_bf16(
        /*neg_a=*/false, a, /*neg_b=*/false, b,
        /*c_mod=*/(short)0, c, /*reuse_a=*/false, /*reuse_b=*/false);
}

// ---------------------------------------------------------------------------
// fp32 -> bf16 elementwise convert
// ---------------------------------------------------------------------------
__global__ void k_cvt_bf16(const float* __restrict__ src,
                           bf16_t* __restrict__ dst, int n) {
    int i = blockIdx.x * blockDim.x + threadIdx.x;
    if (i < n) dst[i] = (bf16_t)src[i];
}

// fp32 [R][C] -> bf16 [C][R]  (weights stored transposed: B-fragment rows
// become 16 contiguous K values -> two b128 loads per fragment)
__global__ void k_transpose_bf16(const float* __restrict__ src,
                                 bf16_t* __restrict__ dst, int R, int C) {
    int i = blockIdx.x * blockDim.x + threadIdx.x;
    if (i < R * C) {
        int r = i / C, c = i - r * C;
        dst[(size_t)c * R + r] = (bf16_t)src[i];
    }
}

// ---------------------------------------------------------------------------
// Generic bf16 WMMA GEMM: C[M][N] = A[M][K] * Bt[N][K]^T  (+ epilogue)
//   MODE 0: out_bf[M][N] = bf16(C)
//   MODE 1: out_vt[B][N][T] = bf16(C)   (transposed store for V)
//   MODE 2: out_bf = bf16(relu(C + bias))          (FFN layer 1)
//   MODE 3: out_f  = C + bias + resid              (FFN layer 2 -> d_out)
// Block: 256 threads = 8 waves arranged 2(M) x 4(N); wave tile 16x64.
// ---------------------------------------------------------------------------
template <int MODE>
__global__ __launch_bounds__(256) void k_gemm_bf16(
    const bf16_t* __restrict__ A, const bf16_t* __restrict__ Bt,
    int M, int N, int K,
    const float* __restrict__ bias, const float* __restrict__ resid,
    bf16_t* __restrict__ outB, float* __restrict__ outF,
    bf16_t* __restrict__ outVt, int Tdim)
{
    const int tid  = threadIdx.x;
    const int wave = tid >> 5, lane = tid & 31;
    const int half = lane >> 4, l16 = lane & 15;
    const int m0 = blockIdx.y * 32 + (wave >> 2) * 16;
    const int n0 = blockIdx.x * 256 + (wave & 3) * 64;

    v8f acc[4];
#pragma unroll
    for (int nt = 0; nt < 4; ++nt) acc[nt] = vzero8();

    const bf16_t* arow = A + (size_t)(m0 + l16) * K;
    const bf16_t* brow[4];
#pragma unroll
    for (int nt = 0; nt < 4; ++nt)
        brow[nt] = Bt + (size_t)(n0 + nt * 16 + l16) * K;

    for (int kk = 0; kk < K; kk += 32) {
        if (kk + 32 < K) {   // warm L1 for the next k-step (no LOADcnt cost)
            __builtin_prefetch(arow + kk + 32 + half * 8, 0, 1);
#pragma unroll
            for (int nt = 0; nt < 4; ++nt)
                __builtin_prefetch(brow[nt] + kk + 32 + half * 16, 0, 1);
        }
        // A fragment (contiguous k order, matches B): k = half*16 + i
        v16bf a = make_frag(ld8(arow + kk + half * 16),
                            ld8(arow + kk + half * 16 + 8));
#pragma unroll
        for (int nt = 0; nt < 4; ++nt) {
            const bf16_t* bp = brow[nt] + kk + half * 16;
            v16bf b = make_frag(ld8(bp), ld8(bp + 8));
            acc[nt] = wmma_bf16(a, b, acc[nt]);
        }
    }

    // Epilogue. D layout: row = m0 + j + 8*half, col = n0 + nt*16 + l16
#pragma unroll
    for (int nt = 0; nt < 4; ++nt) {
        const int col = n0 + nt * 16 + l16;
        const float bv = (MODE == 2 || MODE == 3) ? bias[col] : 0.0f;
#pragma unroll
        for (int j = 0; j < 8; ++j) {
            const int row = m0 + j + 8 * half;
            float v = acc[nt][j] + bv;
            if (MODE == 2 && v < 0.0f) v = 0.0f;
            if (MODE == 0 || MODE == 2) {
                outB[(size_t)row * N + col] = (bf16_t)v;
            } else if (MODE == 1) {
                const int bb = row / Tdim, t = row - bb * Tdim;
                outVt[((size_t)bb * N + col) * Tdim + t] = (bf16_t)v;
            } else { // MODE 3
                outF[(size_t)row * N + col] =
                    v + resid[(size_t)row * N + col];
            }
        }
    }
}

// ---------------------------------------------------------------------------
// Attention: one block = 16 queries of one batch, 8 waves.
// Q tile staged to LDS with the Tensor Data Mover (2D tile 1024x16 bf16).
// Phase 1: waves split keys, online (max, sumexp) per row, combine in LDS.
// Phase 2: per 128-key block, each wave computes one 16x16 score tile,
//          P = exp(s-m)/l -> LDS bf16; then each wave accumulates its
//          128-wide E slice of context = P @ V using Vt[B][E][T].
// ---------------------------------------------------------------------------
__global__ __launch_bounds__(256) void k_attention(
    const bf16_t* __restrict__ Q, const bf16_t* __restrict__ Kmat,
    const bf16_t* __restrict__ Vt, const unsigned char* __restrict__ mask,
    float* __restrict__ ctxF, bf16_t* __restrict__ ctxB)
{
    __shared__ bf16_t Qs[16 * EE];     // 32 KB: this block's Q rows
    __shared__ bf16_t Ps[16 * 128];    // 4 KB : probability tile
    __shared__ float  redM[8 * 16], redL[8 * 16];
    __shared__ float  Mrow[16], Lrow[16];

    const int tid  = threadIdx.x;
    const int wave = tid >> 5, lane = tid & 31;
    const int half = lane >> 4, l16 = lane & 15;
    const int b  = blockIdx.x / (TT / 16);
    const int q0 = (blockIdx.x % (TT / 16)) * 16;
    const float scale = 0.03125f;      // 1/sqrt(1024)

#if USE_TDM
    // ---- TDM: DMA the 16x1024 bf16 Q tile into LDS (wave 0 issues) ----
    if (wave == 0) {
        const unsigned long long ga =
            (unsigned long long)(const char*)(Q + ((size_t)b * TT + q0) * EE);
        const unsigned int lds = (unsigned int)(unsigned long long)(void*)Qs;
        u32x4 g0;
        g0[0] = 1u;                                    // count=1, user D#
        g0[1] = lds;                                   // lds_addr [63:32]
        g0[2] = (unsigned int)(ga & 0xFFFFFFFFu);      // global_addr[31:0]
        g0[3] = (unsigned int)((ga >> 32) & 0x1FFFFFFu) | (2u << 30); // +type=2
        u32x8 g1;
        g1[0] = 1u << 16;          // workgroup_mask=0, data_size=1 (2 bytes)
        g1[1] = (unsigned int)EE << 16;   // tensor_dim0[15:0] @ bits 63:48
        g1[2] = 16u << 16;                // tensor_dim1[15:0] @ bits 95:80
        g1[3] = (unsigned int)EE << 16;   // tile_dim0 @ bits 127:112
        g1[4] = 16u;                      // tile_dim1 @ bits 143:128
        g1[5] = (unsigned int)EE;         // tensor_dim0_stride[31:0]
        g1[6] = 0u;
        g1[7] = 0u;
        asm volatile("tensor_load_to_lds %0, %1"
                     :: "s"(g0), "s"(g1) : "memory");
        __builtin_amdgcn_s_wait_tensorcnt(0);
    }
#else
    for (int i = tid; i < 16 * EE; i += 256)
        Qs[i] = Q[((size_t)b * TT + q0 + (i >> 10)) * EE + (i & (EE - 1))];
#endif
    __syncthreads();

    // ---- Phase 1: softmax stats ----
    float mj[8], lj[8];
#pragma unroll
    for (int j = 0; j < 8; ++j) { mj[j] = -3.0e38f; lj[j] = 0.0f; }

    for (int kt = wave; kt < TT / 16; kt += 8) {
        const int key0 = kt * 16;
        v8f s = vzero8();
        for (int kk = 0; kk < EE; kk += 32) {
            v16bf a = make_frag(ld8(&Qs[l16 * EE + kk + half * 16]),
                                ld8(&Qs[l16 * EE + kk + half * 16 + 8]));
            const bf16_t* bp =
                Kmat + ((size_t)b * TT + key0 + l16) * EE + kk + half * 16;
            v16bf bb = make_frag(ld8(bp), ld8(bp + 8));
            s = wmma_bf16(a, bb, s);
        }
        const int key = key0 + l16;
#pragma unroll
        for (int j = 0; j < 8; ++j) {
            const int q = q0 + j + 8 * half;
            const unsigned char mk = mask[((size_t)b * TT + q) * TT + key];
            const float sv = mk ? -1.0e9f : s[j] * scale;
            float tm = sv;                          // row max over 16 lanes
#pragma unroll
            for (int d = 1; d < 16; d <<= 1) tm = fmaxf(tm, __shfl_xor(tm, d, 32));
            const float nm = fmaxf(mj[j], tm);
            float ts = __expf(sv - nm);             // row sum-exp
#pragma unroll
            for (int d = 1; d < 16; d <<= 1) ts += __shfl_xor(ts, d, 32);
            lj[j] = lj[j] * __expf(mj[j] - nm) + ts;
            mj[j] = nm;
        }
    }
    if (l16 == 0) {
#pragma unroll
        for (int j = 0; j < 8; ++j) {
            redM[wave * 16 + half * 8 + j] = mj[j];
            redL[wave * 16 + half * 8 + j] = lj[j];
        }
    }
    __syncthreads();
    if (tid < 16) {
        float gm = -3.0e38f;
        for (int w = 0; w < 8; ++w) gm = fmaxf(gm, redM[w * 16 + tid]);
        float gl = 0.0f;
        for (int w = 0; w < 8; ++w)
            gl += redL[w * 16 + tid] * __expf(redM[w * 16 + tid] - gm);
        Mrow[tid] = gm;
        Lrow[tid] = (gl > 0.0f) ? gl : 1.0f;
    }
    __syncthreads();

    // ---- Phase 2: context = P @ V, wave owns E slice [e0, e0+128) ----
    const int e0 = wave * 128;
    v8f cacc[8];
#pragma unroll
    for (int nt = 0; nt < 8; ++nt) cacc[nt] = vzero8();

    for (int kb = 0; kb < TT; kb += 128) {
        const int key0 = kb + wave * 16;
        v8f s = vzero8();
        for (int kk = 0; kk < EE; kk += 32) {
            v16bf a = make_frag(ld8(&Qs[l16 * EE + kk + half * 16]),
                                ld8(&Qs[l16 * EE + kk + half * 16 + 8]));
            const bf16_t* bp =
                Kmat + ((size_t)b * TT + key0 + l16) * EE + kk + half * 16;
            v16bf bb = make_frag(ld8(bp), ld8(bp + 8));
            s = wmma_bf16(a, bb, s);
        }
        __syncthreads();   // previous iteration's Ps readers are done
        const int key = key0 + l16;
#pragma unroll
        for (int j = 0; j < 8; ++j) {
            const int r = j + 8 * half;
            const unsigned char mk =
                mask[((size_t)b * TT + q0 + r) * TT + key];
            const float sv = mk ? -1.0e9f : s[j] * scale;
            const float p = __expf(sv - Mrow[r]) / Lrow[r];
            Ps[r * 128 + wave * 16 + l16] = (bf16_t)p;
        }
        __syncthreads();
#pragma unroll
        for (int kk2 = 0; kk2 < 4; ++kk2) {
            v16bf a = make_frag(ld8(&Ps[l16 * 128 + kk2 * 32 + half * 16]),
                                ld8(&Ps[l16 * 128 + kk2 * 32 + half * 16 + 8]));
#pragma unroll
            for (int nt = 0; nt < 8; ++nt) {
                const bf16_t* bp =
                    Vt + ((size_t)b * EE + e0 + nt * 16 + l16) * TT
                       + kb + kk2 * 32 + half * 16;
                v16bf bb = make_frag(ld8(bp), ld8(bp + 8));
                cacc[nt] = wmma_bf16(a, bb, cacc[nt]);
            }
        }
    }

#pragma unroll
    for (int nt = 0; nt < 8; ++nt) {
        const int col = e0 + nt * 16 + l16;
#pragma unroll
        for (int j = 0; j < 8; ++j) {
            const size_t row = (size_t)b * TT + q0 + j + 8 * half;
            const float v = cacc[nt][j];
            ctxF[row * EE + col] = v;
            ctxB[row * EE + col] = (bf16_t)v;
        }
    }
}

// ---------------------------------------------------------------------------
// Host launcher
// ---------------------------------------------------------------------------
extern "C" void kernel_launch(void* const* d_in, const int* in_sizes, int n_in,
                              void* d_out, int out_size, void* d_ws, size_t ws_size,
                              hipStream_t stream) {
    const float* X  = (const float*)d_in[0];
    const float* WQ = (const float*)d_in[1];
    const float* WK = (const float*)d_in[2];
    const float* WV = (const float*)d_in[3];
    const float* W1 = (const float*)d_in[4];
    const float* b1 = (const float*)d_in[5];
    const float* W2 = (const float*)d_in[6];
    const float* b2 = (const float*)d_in[7];
    const unsigned char* mask = (const unsigned char*)d_in[8];
    float* out = (float*)d_out;

    char* ws = (char*)d_ws;
    const size_t MB = 1ull << 20;
    // [0,128MB): Xbf/Qbf/Kbf/Vt during attention stage, aliased by Hbf (FFN)
    bf16_t* Xbf = (bf16_t*)(ws + 0);
    bf16_t* Qbf = (bf16_t*)(ws + 32 * MB);
    bf16_t* Kbf = (bf16_t*)(ws + 64 * MB);
    bf16_t* Vt  = (bf16_t*)(ws + 96 * MB);
    bf16_t* Hbf = (bf16_t*)(ws + 0);            // [M][H] bf16, 128MB
    float*  ctxF = (float*)(ws + 128 * MB);     // 64MB
    bf16_t* ctxB = (bf16_t*)(ws + 192 * MB);    // 32MB
    bf16_t* WqT = (bf16_t*)(ws + 224 * MB);
    bf16_t* WkT = (bf16_t*)(ws + 226 * MB);
    bf16_t* WvT = (bf16_t*)(ws + 228 * MB);
    bf16_t* W1T = (bf16_t*)(ws + 230 * MB);     // [H][E] 8MB
    bf16_t* W2T = (bf16_t*)(ws + 238 * MB);     // [E][H] 8MB, ends 246MB

    // 1) precision conversion + weight transposition
    const int nX = MM * EE;
    k_cvt_bf16<<<(nX + 255) / 256, 256, 0, stream>>>(X, Xbf, nX);
    k_transpose_bf16<<<(EE * EE + 255) / 256, 256, 0, stream>>>(WQ, WqT, EE, EE);
    k_transpose_bf16<<<(EE * EE + 255) / 256, 256, 0, stream>>>(WK, WkT, EE, EE);
    k_transpose_bf16<<<(EE * EE + 255) / 256, 256, 0, stream>>>(WV, WvT, EE, EE);
    k_transpose_bf16<<<(EE * HH + 255) / 256, 256, 0, stream>>>(W1, W1T, EE, HH);
    k_transpose_bf16<<<(HH * EE + 255) / 256, 256, 0, stream>>>(W2, W2T, HH, EE);

    // 2) QKV projections (V stored transposed [B][E][T] for attention B-frags)
    dim3 gp(EE / 256, MM / 32);
    k_gemm_bf16<0><<<gp, 256, 0, stream>>>(Xbf, WqT, MM, EE, EE,
        nullptr, nullptr, Qbf, nullptr, nullptr, TT);
    k_gemm_bf16<0><<<gp, 256, 0, stream>>>(Xbf, WkT, MM, EE, EE,
        nullptr, nullptr, Kbf, nullptr, nullptr, TT);
    k_gemm_bf16<1><<<gp, 256, 0, stream>>>(Xbf, WvT, MM, EE, EE,
        nullptr, nullptr, nullptr, nullptr, Vt, TT);

    // 3) attention -> context (f32 for residual, bf16 for FFN input)
    k_attention<<<BB * (TT / 16), 256, 0, stream>>>(Qbf, Kbf, Vt, mask,
                                                    ctxF, ctxB);

    // 4) FFN: h = relu(ctx @ W1 + b1);  out = h @ W2 + b2 + ctx
    dim3 g1(HH / 256, MM / 32);
    k_gemm_bf16<2><<<g1, 256, 0, stream>>>(ctxB, W1T, MM, HH, EE,
        b1, nullptr, Hbf, nullptr, nullptr, TT);
    dim3 g2(EE / 256, MM / 32);
    k_gemm_bf16<3><<<g2, 256, 0, stream>>>(Hbf, W2T, MM, EE, HH,
        b2, ctxF, nullptr, out, nullptr, TT);
}
